// LSTMAttentionDot_13649406066718
// MI455X (gfx1250) — matrix operand order repaired and checked by reading, not compiled, not run on hardware
//
#include <hip/hip_runtime.h>
#include <hip/hip_bf16.h>

typedef __attribute__((ext_vector_type(16))) _Float16 v16h;
typedef __attribute__((ext_vector_type(8)))  _Float16 v8h;
typedef __attribute__((ext_vector_type(8)))  float    v8f;

static constexpr int BB  = 64;    // batch
static constexpr int TT  = 32;    // time steps
static constexpr int SS  = 128;   // context length
static constexpr int IND = 1024;  // ctx_dim / input size
static constexpr int HH  = 1024;  // hidden size
static constexpr int G4  = 4096;  // 4*H

// ---------------- elementwise helpers ----------------

__global__ void f32_to_f16_kernel(const float* __restrict__ src,
                                  _Float16* __restrict__ dst, int n) {
  int i = blockIdx.x * blockDim.x + threadIdx.x;
  if (i < n) dst[i] = (_Float16)src[i];
}

__global__ void bias_sum_kernel(const float* __restrict__ a,
                                const float* __restrict__ b,
                                float* __restrict__ o, int n) {
  int i = blockIdx.x * blockDim.x + threadIdx.x;
  if (i < n) o[i] = a[i] + b[i];
}

__global__ void copy_f32_kernel(const float* __restrict__ s,
                                float* __restrict__ d, int n) {
  int i = blockIdx.x * blockDim.x + threadIdx.x;
  if (i < n) d[i] = s[i];
}

// hT[b, j] = output[b, T-1, j]
__global__ void gather_hT_kernel(const float* __restrict__ out,
                                 float* __restrict__ hT) {
  int i = blockIdx.x * blockDim.x + threadIdx.x;  // over B*H
  int b = i >> 10, j = i & (HH - 1);
  hT[i] = out[((size_t)b * TT + (TT - 1)) * HH + j];
}

// ---------------- WMMA GEMM: C = act(A * B^T + bias) ----------------
// A: (M x K) f16 row-major, lda in elements
// Bw: (N x K) f16 row-major (i.e. we compute A @ Bw^T), ldb in elements
// One wave32 computes a 16x64 strip of C: the A fragment is loaded once per
// K-chunk and reused across 4 v_wmma_f32_16x16x32_f16 (4 N-tiles).
// All addressing is uniform-base + 32-bit lane offset so the backend can use
// the SADDR global-load form (scalar base, v_offset) instead of per-lane
// 64-bit pointer arithmetic.
__global__ void gemm_wmma_kernel(const _Float16* __restrict__ A, int lda,
                                 const _Float16* __restrict__ Bw, int ldb,
                                 const float* __restrict__ bias,
                                 float* __restrict__ Cf, int ldcf,
                                 _Float16* __restrict__ Ch, int ldch,
                                 int Ngroups, int K, int act, int total) {
  int wave = (blockIdx.x * blockDim.x + threadIdx.x) >> 5;   // wave-uniform
  if (wave >= total) return;   // whole-wave exit: EXEC stays all-1s for WMMA
  int mt = wave / Ngroups;
  int ng = wave - mt * Ngroups;          // group of 4 consecutive N-tiles
  int lane = threadIdx.x & 31;
  int r16  = lane & 15;      // row-within-tile for A, col-within-tile for B/C
  int hf   = lane >> 4;      // which half of the wave

  // A operand order used here (16x32 tile, same k order as B): lane L holds row M=L%16;
  // v16h elem e -> K = hf*16 + e
  int aoff = (mt * 16 + r16) * lda + hf * 16;
  // B operand order used here (32x16 tile, same k order as A): lane L holds col N=L%16;
  // v16h elem e -> K = hf*16 + e  (one contiguous 32B load per lane)
  int boff0 = (ng * 64 + r16) * ldb + hf * 16;
  int boff1 = boff0 + 16 * ldb;
  int boff2 = boff0 + 32 * ldb;
  int boff3 = boff0 + 48 * ldb;

  v8f acc[4];
#pragma unroll
  for (int q = 0; q < 4; ++q) acc[q] = (v8f){0.f, 0.f, 0.f, 0.f, 0.f, 0.f, 0.f, 0.f};

#pragma unroll 2
  for (int k = 0; k < K; k += 32) {
    v8h  alo = *(const v8h*)(A + (aoff + k));
    v8h  ahi = *(const v8h*)(A + (aoff + k + 8));
    v16h b0  = *(const v16h*)(Bw + (boff0 + k));
    v16h b1  = *(const v16h*)(Bw + (boff1 + k));
    v16h b2  = *(const v16h*)(Bw + (boff2 + k));
    v16h b3  = *(const v16h*)(Bw + (boff3 + k));
    if (k + 64 < K) {  // L2-resident working set; keep the pipe warm
      __builtin_prefetch(A  + (aoff  + k + 64), 0, 3);
      __builtin_prefetch(Bw + (boff0 + k + 64), 0, 3);
      __builtin_prefetch(Bw + (boff2 + k + 64), 0, 3);
    }
    v16h av;
#pragma unroll
    for (int e = 0; e < 8; ++e) { av[e] = alo[e]; av[8 + e] = ahi[e]; }
    acc[0] = __builtin_amdgcn_wmma_f32_16x16x32_f16(false, av, false, b0,
                                                    (short)0, acc[0], false, false);
    acc[1] = __builtin_amdgcn_wmma_f32_16x16x32_f16(false, av, false, b1,
                                                    (short)0, acc[1], false, false);
    acc[2] = __builtin_amdgcn_wmma_f32_16x16x32_f16(false, av, false, b2,
                                                    (short)0, acc[2], false, false);
    acc[3] = __builtin_amdgcn_wmma_f32_16x16x32_f16(false, av, false, b3,
                                                    (short)0, acc[3], false, false);
  }

  // C layout: VGPR r -> row (hf*8 + r), col = lane%16
  int crow = mt * 16 + hf * 8;
#pragma unroll
  for (int q = 0; q < 4; ++q) {
    int ccol = ng * 64 + q * 16 + r16;
    float badd = bias ? bias[ccol] : 0.f;
#pragma unroll
    for (int r = 0; r < 8; ++r) {
      float v = acc[q][r] + badd;
      if (act) v = tanhf(v);
      if (Cf) Cf[(size_t)(crow + r) * ldcf + ccol] = v;
      if (Ch) Ch[(size_t)(crow + r) * ldch + ccol] = (_Float16)v;
    }
  }
}

// ---------------- LSTM pointwise cell ----------------
// gates = xg[b*T+t] + hg[b];  i,f,g,o split;  updates c (f32) and hy (f16)
__global__ void lstm_cell_kernel(const float* __restrict__ xg,
                                 const float* __restrict__ hg,
                                 float* __restrict__ c,
                                 _Float16* __restrict__ hy16, int t) {
  int idx = blockIdx.x * blockDim.x + threadIdx.x;  // over B*H
  int b = idx >> 10, j = idx & (HH - 1);
  const float* xr = xg + ((size_t)b * TT + t) * G4;
  const float* hr = hg + (size_t)b * G4;
  float gi = xr[j]          + hr[j];
  float gf = xr[HH + j]     + hr[HH + j];
  float gg = xr[2 * HH + j] + hr[2 * HH + j];
  float go = xr[3 * HH + j] + hr[3 * HH + j];
  float si = 1.f / (1.f + __expf(-gi));
  float sf = 1.f / (1.f + __expf(-gf));
  float so = 1.f / (1.f + __expf(-go));
  float cy = sf * c[idx] + si * tanhf(gg);
  float hy = so * tanhf(cy);
  c[idx] = cy;
  hy16[idx] = (_Float16)hy;
}

// ---------------- attention: scores -> softmax -> weighted ctx ----------------
// One block per batch element. Writes wctx as f16 into concat16[b, 0:IND]
// (inp f16 half concat16[b, IND:2*IND] is written by the inp-GEMM epilogue).
__global__ void attn_kernel(const float* __restrict__ dctx,
                            const float* __restrict__ inp,
                            const float* __restrict__ mask,
                            _Float16* __restrict__ concat16) {
  __shared__ float s_inp[IND];
  __shared__ float s_part[2 * SS];
  __shared__ float s_a[SS];
  __shared__ float red[128];

  int b = blockIdx.x, tid = threadIdx.x;
  const float* dcb = dctx + (size_t)b * SS * IND;

  for (int i = tid; i < IND; i += 256) s_inp[i] = inp[b * IND + i];
  __syncthreads();

  // partial dot products: 2 threads per context row, 512 channels each
  int s = tid & (SS - 1), part = tid >> 7;
  {
    const float* row = dcb + (size_t)s * IND + part * 512;
    const float* iv  = s_inp + part * 512;
    float acc = 0.f;
    for (int c4 = 0; c4 < 512; c4 += 4) {
      float4 dv = *(const float4*)(row + c4);
      acc += dv.x * iv[c4] + dv.y * iv[c4 + 1] + dv.z * iv[c4 + 2] + dv.w * iv[c4 + 3];
    }
    s_part[part * SS + s] = acc;
  }
  __syncthreads();

  float sc = 0.f;
  if (tid < SS) {
    sc = s_part[tid] + s_part[SS + tid] - (1.f - mask[b * SS + tid]) * 100000.f;
    red[tid] = sc;
  }
  __syncthreads();
  for (int o2 = 64; o2 > 0; o2 >>= 1) {
    if (tid < o2) red[tid] = fmaxf(red[tid], red[tid + o2]);
    __syncthreads();
  }
  float mx = red[0];
  __syncthreads();
  float ex = 0.f;
  if (tid < SS) { ex = __expf(sc - mx); red[tid] = ex; }
  __syncthreads();
  for (int o2 = 64; o2 > 0; o2 >>= 1) {
    if (tid < o2) red[tid] += red[tid + o2];
    __syncthreads();
  }
  float denom = red[0];
  __syncthreads();
  if (tid < SS) s_a[tid] = ex / denom;
  __syncthreads();

  // wctx: each thread owns 4 channels
  int cbase = tid * 4;
  float4 w = {0.f, 0.f, 0.f, 0.f};
  for (int ss2 = 0; ss2 < SS; ++ss2) {
    float a = s_a[ss2];
    float4 dv = *(const float4*)(dcb + (size_t)ss2 * IND + cbase);
    w.x += a * dv.x; w.y += a * dv.y; w.z += a * dv.z; w.w += a * dv.w;
  }
  _Float16* cr = concat16 + (size_t)b * (2 * IND) + cbase;
  cr[0] = (_Float16)w.x; cr[1] = (_Float16)w.y;
  cr[2] = (_Float16)w.z; cr[3] = (_Float16)w.w;
}

// ---------------- host side ----------------

extern "C" void kernel_launch(void* const* d_in, const int* in_sizes, int n_in,
                              void* d_out, int out_size, void* d_ws, size_t ws_size,
                              hipStream_t stream) {
  const float* target   = (const float*)d_in[0];
  const float* h0       = (const float*)d_in[1];
  const float* c0       = (const float*)d_in[2];
  const float* ctx      = (const float*)d_in[3];
  const float* ctx_mask = (const float*)d_in[4];
  const float* W_ih     = (const float*)d_in[5];
  const float* b_ih     = (const float*)d_in[6];
  const float* W_hh     = (const float*)d_in[7];
  const float* b_hh     = (const float*)d_in[8];
  const float* W_in     = (const float*)d_in[9];
  const float* W_ctx    = (const float*)d_in[10];
  const float* W_out    = (const float*)d_in[11];
  float* out = (float*)d_out;

  char* ws = (char*)d_ws;
  size_t off = 0;
  auto alloc = [&](size_t bytes) -> void* {
    void* p = ws + off;
    off = (off + bytes + 255) & ~(size_t)255;
    return p;
  };

  _Float16* wih16  = (_Float16*)alloc((size_t)G4 * IND * 2);
  _Float16* whh16  = (_Float16*)alloc((size_t)G4 * HH * 2);
  _Float16* win16  = (_Float16*)alloc((size_t)IND * HH * 2);
  _Float16* wctx16 = (_Float16*)alloc((size_t)IND * 2 * IND * 2);
  _Float16* wout16 = (_Float16*)alloc((size_t)HH * 2 * IND * 2);
  _Float16* ctx16  = (_Float16*)alloc((size_t)BB * SS * 2 * IND * 2);
  _Float16* tgt16  = (_Float16*)alloc((size_t)BB * TT * IND * 2);
  _Float16* h16    = (_Float16*)alloc((size_t)BB * HH * 2);
  _Float16* hy16   = (_Float16*)alloc((size_t)BB * HH * 2);
  _Float16* cat16  = (_Float16*)alloc((size_t)BB * 2 * IND * 2);
  float* biasbuf   = (float*)alloc((size_t)G4 * 4);
  float* dctx      = (float*)alloc((size_t)BB * SS * IND * 4);
  float* xg        = (float*)alloc((size_t)BB * TT * G4 * 4);
  float* hg        = (float*)alloc((size_t)BB * G4 * 4);
  float* cbuf      = (float*)alloc((size_t)BB * HH * 4);
  float* inpf      = (float*)alloc((size_t)BB * IND * 4);

  auto cvt = [&](const float* s, _Float16* d, int n) {
    f32_to_f16_kernel<<<(n + 255) / 256, 256, 0, stream>>>(s, d, n);
  };
  auto gemm = [&](const _Float16* A, int lda, const _Float16* Bm, int ldb,
                  const float* bias, float* Cf, int ldcf, _Float16* Ch, int ldch,
                  int M, int N, int K, int act) {
    int tiles = (M / 16) * (N / 64);          // one wave per 16x64 strip
    int blocks = (tiles * 32 + 255) / 256;
    gemm_wmma_kernel<<<blocks, 256, 0, stream>>>(A, lda, Bm, ldb, bias, Cf, ldcf,
                                                 Ch, ldch, N / 64, K, act, tiles);
  };

  // --- one-time conversions / precomputes (weights are L2-resident afterwards) ---
  cvt(W_ih,  wih16,  G4 * IND);
  cvt(W_hh,  whh16,  G4 * HH);
  cvt(W_in,  win16,  IND * HH);
  cvt(W_ctx, wctx16, IND * 2 * IND);
  cvt(W_out, wout16, HH * 2 * IND);
  cvt(ctx,    ctx16, BB * SS * 2 * IND);
  cvt(target, tgt16, BB * TT * IND);
  cvt(h0,     h16,   BB * HH);
  copy_f32_kernel<<<(BB * HH + 255) / 256, 256, 0, stream>>>(c0, cbuf, BB * HH);
  bias_sum_kernel<<<(G4 + 255) / 256, 256, 0, stream>>>(b_ih, b_hh, biasbuf, G4);

  // dense_ctx = ctx @ W_ctx^T   (8192 x 1024 x 2048)
  gemm(ctx16, 2 * IND, wctx16, 2 * IND, nullptr, dctx, IND, nullptr, 0,
       BB * SS, IND, 2 * IND, 0);
  // xg = target @ W_ih^T + (b_ih + b_hh)   (2048 x 4096 x 1024)
  gemm(tgt16, IND, wih16, IND, biasbuf, xg, G4, nullptr, 0,
       BB * TT, G4, IND, 0);

  // --- serial recurrence ---
  for (int t = 0; t < TT; ++t) {
    // hg = h @ W_hh^T   (64 x 4096 x 1024)
    gemm(h16, HH, whh16, HH, nullptr, hg, G4, nullptr, 0, BB, G4, HH, 0);
    // LSTM pointwise: cy, hy
    lstm_cell_kernel<<<(BB * HH + 255) / 256, 256, 0, stream>>>(xg, hg, cbuf, hy16, t);
    // inp = hy @ W_in^T  (64 x 1024 x 1024); f16 epilogue -> concat[:, IND:2*IND]
    gemm(hy16, HH, win16, HH, nullptr, inpf, IND, cat16 + IND, 2 * IND,
         BB, IND, HH, 0);
    // attention -> concat[:, 0:IND]
    attn_kernel<<<BB, 256, 0, stream>>>(dctx, inpf, ctx_mask, cat16);
    // h_tilde = tanh(concat @ W_out^T)  (64 x 1024 x 2048)
    // f32 epilogue -> output[b, t, :] (row stride T*H), f16 -> next h
    gemm(cat16, 2 * IND, wout16, 2 * IND, nullptr, out + (size_t)t * HH, TT * HH,
         h16, HH, BB, HH, 2 * IND, 1);
  }

  // hT and cT tail outputs
  gather_hT_kernel<<<(BB * HH + 255) / 256, 256, 0, stream>>>(
      out, out + (size_t)BB * TT * HH);
  copy_f32_kernel<<<(BB * HH + 255) / 256, 256, 0, stream>>>(
      cbuf, out + (size_t)BB * TT * HH + (size_t)BB * HH, BB * HH);
}
